// GCNlayer_40029095199322
// MI455X (gfx1250) — matrix operand order repaired and checked by reading, not compiled, not run on hardware
//
#include <hip/hip_runtime.h>

// ---------------------------------------------------------------------------
// ARMA-GNN forward for MI455X (gfx1250, wave32, WMMA).
//   - dual-output WMMA GEMM (h@Wi, h@Wr+b), fragment-native LDS layout:
//     operands load as aligned ds_load_b128; B-fragment loads are software-
//     pipelined one N-tile ahead so LDS latency hides under WMMA execution
//   - edge scatter-add with f32 atomics into an L2-resident accumulator
//   - ReLU + BatchNorm affine folded into the next GEMM's A-load / pooling
// ---------------------------------------------------------------------------

#define N_NODES 100000
#define N_EDGES 1600000
#define N_GRAPH 2000
#define FIN     128

#define AS_STRIDE 56   // halfs; 112B row stride: 16B-aligned frags, conflict-free
#define BS_STRIDE 40   // halfs;  80B row stride: 16B-aligned frags, conflict-free

typedef __attribute__((ext_vector_type(16))) _Float16 v16h;
typedef __attribute__((ext_vector_type(8)))  _Float16 v8h;
typedef __attribute__((ext_vector_type(8)))  float    v8f;

// ------------------------------- utility kernels ---------------------------

__global__ void k_zero(float* __restrict__ p, int n) {
  int i = blockIdx.x * blockDim.x + threadIdx.x;
  if (i < n) p[i] = 0.0f;
}

__global__ void k_deg(const int* __restrict__ col, float* __restrict__ deg, int E) {
  int i = blockIdx.x * blockDim.x + threadIdx.x;
  if (i < E) atomicAdd(&deg[col[i]], 1.0f);
}

__global__ void k_dinv(const float* __restrict__ deg, float* __restrict__ dinv, int N) {
  int i = blockIdx.x * blockDim.x + threadIdx.x;
  if (i < N) {
    float d = deg[i];
    dinv[i] = d > 0.0f ? rsqrtf(d) : 0.0f;
  }
}

// ------------------------------- dual WMMA GEMM ----------------------------
// Hout = T(x)@Wi, accOut = T(x)@Wr + bias for a 128-row block and a 128-col
// slice (blockIdx.y).  T(x) = x (mode 0) or relu(x)*scale+shift (mode 1).
// 256 threads = 8 waves; wave w owns rows [16w,16w+16), all 8 N-tiles.

__launch_bounds__(256)
__global__ void k_gemm_dual(const float* __restrict__ xin,
                            const float* __restrict__ scale,
                            const float* __restrict__ shift,
                            int mode,
                            const float* __restrict__ Wi,
                            const float* __restrict__ Wr,
                            const float* __restrict__ bias,
                            float* __restrict__ Hout,
                            float* __restrict__ accOut,
                            int Nrows, int Fout) {
  // A: row-major [128][AS_STRIDE]; fragment at k = hi*16
  // B: K-major per output column [128 n][BS_STRIDE k]; fragment at k = hi*16
  __shared__ __align__(16) _Float16 As [128 * AS_STRIDE];
  __shared__ __align__(16) _Float16 Bsw[2][128 * BS_STRIDE];   // [0]=Wi [1]=Wr

  const int t    = threadIdx.x;
  const int wave = t >> 5;
  const int lane = t & 31;
  const int hi   = lane >> 4;
  const int nl   = lane & 15;
  const int mbase    = wave * 16;
  const int blockRow = blockIdx.x * 128;
  const int colBase  = blockIdx.y * 128;

  v8f acc_i[8] = {};
  v8f acc_r[8] = {};

  for (int ks = 0; ks < 4; ++ks) {          // K = 128 in 4 steps of 32
    // ---- stage A tile (128 rows x 32 k), ReLU+BN affine on load ----
    {
      int r  = t >> 1;
      int k0 = (t & 1) << 4;
      int gr = blockRow + r; if (gr >= Nrows) gr = Nrows - 1;
      const float* src = xin + (size_t)gr * FIN + ks * 32 + k0;
      v8h a0, a1;
      #pragma unroll
      for (int i = 0; i < 8; ++i) {
        float v0 = src[i], v1 = src[8 + i];
        if (mode) {
          int f = ks * 32 + k0 + i;
          v0 = fmaxf(v0, 0.0f) * scale[f]     + shift[f];
          v1 = fmaxf(v1, 0.0f) * scale[f + 8] + shift[f + 8];
        }
        a0[i] = (_Float16)v0;
        a1[i] = (_Float16)v1;
      }
      *(v8h*)&As[r * AS_STRIDE + k0]     = a0;
      *(v8h*)&As[r * AS_STRIDE + k0 + 8] = a1;
    }
    // ---- stage B tiles swizzled K-major (32 k x 128 n, Wi and Wr) ----
    {
      int n  = t >> 1;               // output column within slice
      int kh = (t & 1) << 4;         // which 16-K half
      const float* wsrc[2] = { Wi, Wr };
      #pragma unroll
      for (int m2 = 0; m2 < 2; ++m2) {
        const float* wp = wsrc[m2] + (size_t)(ks * 32 + kh) * Fout + colBase + n;
        if (ks < 3) __builtin_prefetch(wp + 32 * (size_t)Fout, 0, 1);
        v8h b0, b1;
        #pragma unroll
        for (int i = 0; i < 8; ++i) {
          b0[i] = (_Float16)wp[(size_t)i * Fout];
          b1[i] = (_Float16)wp[(size_t)(i + 8) * Fout];
        }
        *(v8h*)&Bsw[m2][n * BS_STRIDE + kh]     = b0;
        *(v8h*)&Bsw[m2][n * BS_STRIDE + kh + 8] = b1;
      }
    }
    __syncthreads();

    // ---- A fragment: two aligned ds_load_b128 ----
    const _Float16* ap = &As[(mbase + nl) * AS_STRIDE + hi * 16];
    v8h a0 = *(const v8h*)(ap);
    v8h a1 = *(const v8h*)(ap + 8);
    v16h a = __builtin_shufflevector(a0, a1, 0,1,2,3,4,5,6,7,8,9,10,11,12,13,14,15);

    // ---- 8 N-tiles x 2 outputs; B-fragment loads pipelined 1 tile ahead ----
    const int fragOff = nl * BS_STRIDE + hi * 16;
    #define LOAD_BFRAG(mat, nt, dst)                                           \
      {                                                                        \
        const _Float16* bp = &Bsw[mat][(nt) * 16 * BS_STRIDE + fragOff];       \
        v8h lo = *(const v8h*)(bp);                                            \
        v8h hi8 = *(const v8h*)(bp + 8);                                       \
        dst = __builtin_shufflevector(lo, hi8,                                 \
                0,1,2,3,4,5,6,7,8,9,10,11,12,13,14,15);                        \
      }

    v16h bi_cur, br_cur, bi_nxt, br_nxt;
    LOAD_BFRAG(0, 0, bi_cur)
    LOAD_BFRAG(1, 0, br_cur)
    #pragma unroll
    for (int nt = 0; nt < 8; ++nt) {
      int ntn = (nt + 1) & 7;                  // tail reload of tile 0: harmless
      LOAD_BFRAG(0, ntn, bi_nxt)
      LOAD_BFRAG(1, ntn, br_nxt)
      acc_i[nt] = __builtin_amdgcn_wmma_f32_16x16x32_f16(
          false, a, false, bi_cur, (short)0, acc_i[nt], false, false);
      acc_r[nt] = __builtin_amdgcn_wmma_f32_16x16x32_f16(
          false, a, false, br_cur, (short)0, acc_r[nt], false, false);
      bi_cur = bi_nxt;
      br_cur = br_nxt;
    }
    #undef LOAD_BFRAG
    __syncthreads();
  }

  // ---- epilogue: C/D layout M = hi*8+v, N = nl ----
  #pragma unroll
  for (int nt = 0; nt < 8; ++nt) {
    int coln = colBase + nt * 16 + nl;
    float bv = bias[coln];
    #pragma unroll
    for (int v = 0; v < 8; ++v) {
      int row = blockRow + mbase + hi * 8 + v;
      if (row < Nrows) {
        Hout  [(size_t)row * Fout + coln] = acc_i[nt][v];
        accOut[(size_t)row * Fout + coln] = acc_r[nt][v] + bv;
      }
    }
  }
}

// ------------------------------- edge scatter ------------------------------
// One wave per edge; 32 lanes x float4 covers 128 feats (x2 for 256).
// H and acc (51-102MB) are L2-resident (192MB L2): gather + atomic scatter
// traffic never reaches HBM.

__launch_bounds__(256)
__global__ void k_edge(const float* __restrict__ H,
                       const int* __restrict__ row, const int* __restrict__ col,
                       const float* __restrict__ dinv,
                       float* __restrict__ acc, int E, int Fout) {
  int gw   = (blockIdx.x * blockDim.x + threadIdx.x) >> 5;
  int lane = threadIdx.x & 31;
  if (gw >= E) return;
  int r = row[gw], c = col[gw];
  float nrm = dinv[r] * dinv[c];
  const float4* hs = (const float4*)(H + (size_t)r * Fout);
  float* ad = acc + (size_t)c * Fout;
  int nv = Fout >> 7;
  for (int i = 0; i < nv; ++i) {
    int e4 = lane + i * 32;
    float4 v = hs[e4];
    atomicAdd(ad + e4 * 4 + 0, v.x * nrm);
    atomicAdd(ad + e4 * 4 + 1, v.y * nrm);
    atomicAdd(ad + e4 * 4 + 2, v.z * nrm);
    atomicAdd(ad + e4 * 4 + 3, v.w * nrm);
  }
}

// --------------------------- BN stats over relu(acc) -----------------------

__global__ void k_relu_stats(const float* __restrict__ acc, float* __restrict__ stats,
                             int Nrows, int F, int rowsPerBlock) {
  int f  = threadIdx.x;                       // blockDim.x == F
  int r0 = blockIdx.x * rowsPerBlock;
  int r1 = r0 + rowsPerBlock; if (r1 > Nrows) r1 = Nrows;
  float s = 0.0f, sq = 0.0f;
  for (int r = r0; r < r1; ++r) {
    float v = fmaxf(acc[(size_t)r * F + f], 0.0f);
    s += v; sq += v * v;
  }
  atomicAdd(&stats[f], s);
  atomicAdd(&stats[F + f], sq);
}

__global__ void k_bn_finalize(const float* __restrict__ stats,
                              const float* __restrict__ gamma,
                              const float* __restrict__ beta,
                              float* __restrict__ scale, float* __restrict__ shift,
                              int Nrows, int F) {
  int f = threadIdx.x;
  if (f < F) {
    float invN = 1.0f / (float)Nrows;
    float mu   = stats[f] * invN;
    float var  = stats[F + f] * invN - mu * mu;   // biased batch variance
    float sc   = gamma[f] * rsqrtf(var + 1e-5f);
    scale[f] = sc;
    shift[f] = beta[f] - mu * sc;
  }
}

// ------------------------------- pooling -----------------------------------

__global__ void k_pool(const float* __restrict__ acc6, const int* __restrict__ batch,
                       const float* __restrict__ scale, const float* __restrict__ shift,
                       float* __restrict__ out, int Nrows, int F) {
  int idx = blockIdx.x * blockDim.x + threadIdx.x;
  if (idx < Nrows * F) {
    int node = idx / F;
    int f    = idx - node * F;
    float v = fmaxf(acc6[idx], 0.0f) * scale[f] + shift[f];
    atomicAdd(&out[(size_t)batch[node] * F + f], v);
  }
}

// ------------------------------- launcher ----------------------------------

extern "C" void kernel_launch(void* const* d_in, const int* in_sizes, int n_in,
                              void* d_out, int out_size, void* d_ws, size_t ws_size,
                              hipStream_t stream) {
  (void)in_sizes; (void)n_in; (void)out_size; (void)ws_size;

  const float* x   = (const float*)d_in[0];
  const int*   ei  = (const int*)  d_in[1];
  const int*   bat = (const int*)  d_in[2];
  const float* Wi  = (const float*)d_in[3];
  const float* Wr  = (const float*)d_in[4];
  const float* b   = (const float*)d_in[5];
  const float* g   = (const float*)d_in[6];
  const float* be  = (const float*)d_in[7];
  const float* Wi6 = (const float*)d_in[8];
  const float* Wr6 = (const float*)d_in[9];
  const float* b6  = (const float*)d_in[10];
  const float* g6  = (const float*)d_in[11];
  const float* be6 = (const float*)d_in[12];

  const int N = N_NODES, E = N_EDGES, G = N_GRAPH;
  const int* row = ei;
  const int* col = ei + E;

  float* ws = (float*)d_ws;
  size_t off = 0;
  float* deg   = ws + off; off += N;
  float* dinv  = ws + off; off += N;
  float* stats = ws + off; off += 512;
  float* scale = ws + off; off += 256;
  float* shift = ws + off; off += 256;
  off = (off + 255) & ~(size_t)255;
  float* accA = ws + off; off += (size_t)N * 128;
  float* accB = ws + off; off += (size_t)N * 128;
  float* Hbuf = ws + off; off += (size_t)N * 256;   // N x 128 (l<5) / N x 256 (l=6)
  float* acc6 = ws + off; off += (size_t)N * 256;

  float* out = (float*)d_out;

  auto cdiv = [](int a, int d) { return (a + d - 1) / d; };
  dim3 b256(256);

  k_zero<<<cdiv(N, 256),     b256, 0, stream>>>(deg, N);
  k_zero<<<cdiv(G*256, 256), b256, 0, stream>>>(out, G * 256);
  k_deg <<<cdiv(E, 256),     b256, 0, stream>>>(col, deg, E);
  k_dinv<<<cdiv(N, 256),     b256, 0, stream>>>(deg, dinv, N);

  const int gemmBlocks = cdiv(N, 128);
  const int edgeBlocks = cdiv(E, 8);        // 8 waves/block, 1 wave/edge

  float* prev = nullptr;
  for (int l = 0; l < 5; ++l) {
    const float* in = (l == 0) ? x : prev;
    int mode = (l == 0) ? 0 : 1;
    float* cur = (l & 1) ? accB : accA;

    k_gemm_dual<<<dim3(gemmBlocks, 1), b256, 0, stream>>>(
        in, scale, shift, mode,
        Wi + (size_t)l * 128 * 128, Wr + (size_t)l * 128 * 128, b + l * 128,
        Hbuf, cur, N, 128);
    k_edge<<<edgeBlocks, b256, 0, stream>>>(Hbuf, row, col, dinv, cur, E, 128);
    k_zero<<<1, b256, 0, stream>>>(stats, 256);
    k_relu_stats<<<cdiv(N, 512), dim3(128), 0, stream>>>(cur, stats, N, 128, 512);
    k_bn_finalize<<<1, dim3(128), 0, stream>>>(stats, g + l * 128, be + l * 128,
                                               scale, shift, N, 128);
    prev = cur;
  }

  // layer 6: 128 -> 256 (grid.y = 2 column slices)
  k_gemm_dual<<<dim3(gemmBlocks, 2), b256, 0, stream>>>(
      prev, scale, shift, 1, Wi6, Wr6, b6, Hbuf, acc6, N, 256);
  k_edge<<<edgeBlocks, b256, 0, stream>>>(Hbuf, row, col, dinv, acc6, E, 256);
  k_zero<<<2, b256, 0, stream>>>(stats, 512);
  k_relu_stats<<<cdiv(N, 512), dim3(256), 0, stream>>>(acc6, stats, N, 256, 512);
  k_bn_finalize<<<1, dim3(256), 0, stream>>>(stats, g6, be6, scale, shift, N, 256);
  k_pool<<<cdiv(N * 256, 256), b256, 0, stream>>>(acc6, bat, scale, shift, out, N, 256);
}
